// DiTBlock_89970974917091
// MI455X (gfx1250) — matrix operand order repaired and checked by reading, not compiled, not run on hardware
//
#include <hip/hip_runtime.h>
#include <math.h>

// Problem constants (fixed by the harness/reference)
#define N_NODES 40000
#define N_EDGES 480000
#define DMODEL 128
#define HEADS 8
#define HD 16
#define RELD 64
#define EDD 32
#define MLPH 512

typedef __attribute__((ext_vector_type(16))) __bf16 v16bf;
typedef __attribute__((ext_vector_type(8)))  float  v8f;
typedef __attribute__((ext_vector_type(4)))  unsigned int uint4v;

union ABf { v16bf v; uint4v q[2]; unsigned short u[16]; };
union CF  { v8f v; float f[8]; };

__device__ __forceinline__ unsigned short f2bf(float f) {
  unsigned u = __float_as_uint(f);
  u += 0x7FFFu + ((u >> 16) & 1u);  // round-to-nearest-even
  return (unsigned short)(u >> 16);
}
__device__ __forceinline__ unsigned pk2(float a, float b) {
  return (unsigned)f2bf(a) | ((unsigned)f2bf(b) << 16);
}
__device__ __forceinline__ float bf2f(unsigned short h) {
  return __uint_as_float(((unsigned)h) << 16);
}
__device__ __forceinline__ float siluf(float x) { return x / (1.f + __expf(-x)); }

// CDNA5 packed bf16 atomic add (GLOBAL_ATOMIC_PK_ADD_BF16, no return)
__device__ __forceinline__ void atomic_pk_add_bf16(unsigned short* p, unsigned data) {
  unsigned long long a = (unsigned long long)p;
  asm volatile("global_atomic_pk_add_bf16 %0, %1, off" ::"v"(a), "v"(data)
               : "memory");
}

// D = A(16x32 bf16) * B(32x16 bf16) + C(16x16 f32) -- v_wmma_f32_16x16x32_bf16
__device__ __forceinline__ v8f wmma_bf16(v16bf a, v16bf b, v8f c) {
  return __builtin_amdgcn_wmma_f32_16x16x32_bf16(false, a, false, b, (short)0, c,
                                                 false, false);
}

// A fragment from LDS (row-major [16][ld] bf16). ISA layout per lane:
// row M = lane&15; elems 0..7 -> K=kh+0..7, elems 8..15 -> K=kh+16..23
// (kh = 8 for lanes 16..31). Two contiguous 16B runs -> two ds_load_b128.
__device__ __forceinline__ v16bf ld_a_lds(const unsigned short* base, int ld,
                                          int lane, int kbase) {
  ABf r;
  const unsigned short* p =
      base + (lane & 15) * ld + kbase + ((lane & 16) ? 8 : 0);
  r.q[0] = *(const uint4v*)(p);
  r.q[1] = *(const uint4v*)(p + 16);
  return r.v;
}
// Same fragment layout but sourced from global row-major bf16 (e.g. agg)
__device__ __forceinline__ v16bf ld_a_glb(const unsigned short* base, int ld,
                                          int lane, int kbase) {
  ABf r;
  const unsigned short* p =
      base + (size_t)(lane & 15) * ld + kbase + ((lane & 16) ? 8 : 0);
  r.q[0] = *(const uint4v*)(p);
  r.q[1] = *(const uint4v*)(p + 16);
  return r.v;
}

// B fragment from fragment-major packed weights: tile (kt,nt) is 512 bf16,
// lane's 16 values contiguous -> two global_load_b128, wave reads 1KB coalesced.
__device__ __forceinline__ v16bf ld_b_pk(const unsigned short* wpk, int ntiles,
                                         int lane, int kt, int nt) {
  ABf r;
  const uint4v* p = (const uint4v*)(wpk + (((size_t)kt * ntiles + nt) << 9) +
                                    (lane << 4));
  r.q[0] = p[0];
  r.q[1] = p[1];
  return r.v;
}

// ---------------- weight prep: f32 -> bf16 fragment-major tiles --------------
__global__ void k_pack_b(const float* __restrict__ W, unsigned short* __restrict__ d,
                         int K, int N) {
  int i = blockIdx.x * blockDim.x + threadIdx.x;
  if (i >= K * N) return;
  int tile = i >> 9, rem = i & 511, lane = rem >> 4, e = rem & 15;
  int ntiles = N >> 4;
  int kt = tile / ntiles, nt = tile - kt * ntiles;
  int k = kt * 32 + ((e < 8) ? e : e + 8) + ((lane & 16) ? 8 : 0);
  int n = nt * 16 + (lane & 15);
  d[i] = f2bf(W[(size_t)k * N + n]);
}
// pack Wbias(32x8) | Wgate(32x8) -> one fragment-major 32x16 tile
__global__ void k_pack_bias_gate(const float* __restrict__ Wb,
                                 const float* __restrict__ Wg,
                                 unsigned short* __restrict__ d) {
  int i = blockIdx.x * blockDim.x + threadIdx.x;
  if (i < 512) {
    int lane = i >> 4, e = i & 15;
    int k = ((e < 8) ? e : e + 8) + ((lane & 16) ? 8 : 0);
    int n = lane & 15;
    float v = (n < 8) ? Wb[k * HEADS + n] : Wg[k * HEADS + (n - 8)];
    d[i] = f2bf(v);
  }
}
__global__ void k_init(unsigned* aggz, float* denom, int* segmax) {
  int i = blockIdx.x * blockDim.x + threadIdx.x;
  if (i < N_NODES * (DMODEL / 2)) aggz[i] = 0u;  // bf16 pairs
  if (i < N_NODES * HEADS) { denom[i] = 0.f; segmax[i] = (int)0x80000000u; }
}

// ---------------- pass 1: per-16-node tile: ada[0:256], h, q/k/v, u ----------
__global__ __launch_bounds__(256) void k_node_pre(
    const float* __restrict__ x, const float* __restrict__ cin,
    const unsigned short* __restrict__ wWada, const float* __restrict__ bada,
    const unsigned short* __restrict__ wWq, const float* __restrict__ bq,
    const unsigned short* __restrict__ wWk, const float* __restrict__ bk,
    const unsigned short* __restrict__ wWv, const float* __restrict__ bv,
    const unsigned short* __restrict__ wWrel, const float* __restrict__ brel,
    const float* __restrict__ g_reln, const float* __restrict__ b_reln,
    float* __restrict__ q, float* __restrict__ k, float* __restrict__ v,
    float* __restrict__ u) {
  extern __shared__ char smem[];
  float* lnx          = (float*)smem;                          // 16*128 f32
  unsigned short* sAc = (unsigned short*)(lnx + 16 * DMODEL);  // 16*128 bf16
  float* smod         = (float*)(sAc + 16 * DMODEL);           // 16*256 f32
  unsigned short* hA  = (unsigned short*)(smod + 16 * 256);    // 16*128 bf16
  float* relbuf       = (float*)(hA + 16 * DMODEL);            // 16*64  f32
  float* red          = relbuf + 16 * RELD;                    // 512    f32

  const int tid = threadIdx.x, lane = tid & 31, wave = tid >> 5;
  const int node0 = blockIdx.x * 16;
  const int row = tid >> 4, sub = tid & 15;  // 16 threads per row, 8 cols each

  // LN(x) stats + silu(c) staging (vectorized, x kept in registers)
  {
    const float* xr = x + (size_t)(node0 + row) * DMODEL + sub * 8;
    const float* cr = cin + (size_t)(node0 + row) * DMODEL + sub * 8;
    float4 xv[2];
    float s = 0.f, s2 = 0.f;
#pragma unroll
    for (int c4 = 0; c4 < 2; ++c4) {
      xv[c4] = *(const float4*)(xr + c4 * 4);
      float4 cv = *(const float4*)(cr + c4 * 4);
      unsigned* dp = (unsigned*)(sAc + row * DMODEL + sub * 8 + c4 * 4);
      dp[0] = pk2(siluf(cv.x), siluf(cv.y));
      dp[1] = pk2(siluf(cv.z), siluf(cv.w));
      s += xv[c4].x + xv[c4].y + xv[c4].z + xv[c4].w;
      s2 += xv[c4].x * xv[c4].x + xv[c4].y * xv[c4].y + xv[c4].z * xv[c4].z +
            xv[c4].w * xv[c4].w;
    }
    red[row * 16 + sub] = s; red[256 + row * 16 + sub] = s2;
    __syncthreads();
    float su = 0.f, sq = 0.f;
    for (int t = 0; t < 16; ++t) { su += red[row * 16 + t]; sq += red[256 + row * 16 + t]; }
    float mu = su / DMODEL, var = sq / DMODEL - mu * mu;
    float inv = rsqrtf(var + 1e-6f);
#pragma unroll
    for (int c4 = 0; c4 < 2; ++c4) {
      float4 t = xv[c4];
      t.x = (t.x - mu) * inv; t.y = (t.y - mu) * inv;
      t.z = (t.z - mu) * inv; t.w = (t.w - mu) * inv;
      *(float4*)(lnx + row * DMODEL + sub * 8 + c4 * 4) = t;
    }
  }
  __syncthreads();

  // ada columns 0..255 (sh_msa | sc_msa): 16 n-tiles, K=128
  {
    v16bf af[4];
#pragma unroll
    for (int ks = 0; ks < 4; ++ks) af[ks] = ld_a_lds(sAc, DMODEL, lane, ks * 32);
    for (int nt = wave; nt < 16; nt += 8) {
      if (nt + 8 < 16)  // prefetch next n-tile's first B tile
        __builtin_prefetch(wWada + (((size_t)0 * 48 + nt + 8) << 9) + (lane << 4), 0, 2);
      CF acc; float bb = bada[nt * 16 + (lane & 15)];
#pragma unroll
      for (int g = 0; g < 8; ++g) acc.f[g] = bb;
#pragma unroll
      for (int ks = 0; ks < 4; ++ks) {
        v16bf bf = ld_b_pk(wWada, 48, lane, ks, nt);
        acc.v = wmma_bf16(af[ks], bf, acc.v);
      }
      int crow = (lane & 16) ? 8 : 0, col = nt * 16 + (lane & 15);
#pragma unroll
      for (int g = 0; g < 8; ++g) smod[(g + crow) * 256 + col] = acc.f[g];
    }
  }
  __syncthreads();

  // h = ln(x)*(1+sc_msa)+sh_msa -> bf16
  for (int i = tid; i < 16 * 32; i += 256) {
    int r = i >> 5, cb = (i & 31) * 4;
    float4 lv = *(const float4*)(lnx + r * DMODEL + cb);
    float4 sh = *(const float4*)(smod + r * 256 + cb);
    float4 sc = *(const float4*)(smod + r * 256 + 128 + cb);
    unsigned* dp = (unsigned*)(hA + r * DMODEL + cb);
    dp[0] = pk2(lv.x * (1.f + sc.x) + sh.x, lv.y * (1.f + sc.y) + sh.y);
    dp[1] = pk2(lv.z * (1.f + sc.z) + sh.z, lv.w * (1.f + sc.w) + sh.w);
  }
  __syncthreads();

  // q/k/v (8 n-tiles each) and rel (4 n-tiles), K=128
  {
    v16bf af[4];
#pragma unroll
    for (int ks = 0; ks < 4; ++ks) af[ks] = ld_a_lds(hA, DMODEL, lane, ks * 32);
    for (int t = wave; t < 28; t += 8) {
      const unsigned short* W; const float* bias; int nt, which;
      if (t < 8)       { W = wWq;   bias = bq;   nt = t;      which = 0; }
      else if (t < 16) { W = wWk;   bias = bk;   nt = t - 8;  which = 1; }
      else if (t < 24) { W = wWv;   bias = bv;   nt = t - 16; which = 2; }
      else             { W = wWrel; bias = brel; nt = t - 24; which = 3; }
      int ntiles = (which == 3) ? 4 : 8;
      CF acc; float bb = bias[nt * 16 + (lane & 15)];
#pragma unroll
      for (int g = 0; g < 8; ++g) acc.f[g] = bb;
#pragma unroll
      for (int ks = 0; ks < 4; ++ks) {
        v16bf bf = ld_b_pk(W, ntiles, lane, ks, nt);
        acc.v = wmma_bf16(af[ks], bf, acc.v);
      }
      int crow = (lane & 16) ? 8 : 0, col = nt * 16 + (lane & 15);
      if (which == 3) {
#pragma unroll
        for (int g = 0; g < 8; ++g) relbuf[(g + crow) * RELD + col] = acc.f[g];
      } else {
        float* dst = (which == 0) ? q : (which == 1) ? k : v;
#pragma unroll
        for (int g = 0; g < 8; ++g)
          dst[(size_t)(node0 + g + crow) * DMODEL + col] = acc.f[g];
      }
    }
  }
  __syncthreads();

  // u = LN(rel)*g_reln + b_reln   (4 cols per thread, vectorized)
  {
    float4 rv = *(const float4*)(relbuf + row * RELD + sub * 4);
    float s = rv.x + rv.y + rv.z + rv.w;
    float s2 = rv.x * rv.x + rv.y * rv.y + rv.z * rv.z + rv.w * rv.w;
    red[row * 16 + sub] = s; red[256 + row * 16 + sub] = s2;
    __syncthreads();
    float su = 0.f, sq = 0.f;
    for (int t = 0; t < 16; ++t) { su += red[row * 16 + t]; sq += red[256 + row * 16 + t]; }
    float mu = su / RELD, var = sq / RELD - mu * mu, inv = rsqrtf(var + 1e-6f);
    float4 gg = *(const float4*)(g_reln + sub * 4);
    float4 bbv = *(const float4*)(b_reln + sub * 4);
    float4 o;
    o.x = (rv.x - mu) * inv * gg.x + bbv.x;
    o.y = (rv.y - mu) * inv * gg.y + bbv.y;
    o.z = (rv.z - mu) * inv * gg.z + bbv.z;
    o.w = (rv.w - mu) * inv * gg.w + bbv.w;
    *(float4*)(u + (size_t)(node0 + row) * RELD + sub * 4) = o;
  }
}

// ---------------- pass 2: per-16-edge tile edge MLP + logits -----------------
__global__ __launch_bounds__(128) void k_edge(
    const int* __restrict__ edge_index, const float* __restrict__ u,
    const float* __restrict__ q, const float* __restrict__ k,
    const unsigned short* __restrict__ wW1e, const float* __restrict__ b1e,
    const unsigned short* __restrict__ wW2e, const float* __restrict__ b2e,
    const unsigned short* __restrict__ wWbg,
    float* __restrict__ sim, float* __restrict__ gate, int* __restrict__ segmax) {
  extern __shared__ char smem[];
  int* sSrc           = (int*)smem;                        // 4*16
  int* sDst           = sSrc + 4 * 16;                     // 4*16
  unsigned short* ef0 = (unsigned short*)(sDst + 4 * 16);  // 4*16*192
  unsigned short* s1  = ef0 + 4 * 16 * 192;                // 4*16*64
  unsigned short* s2  = s1 + 4 * 16 * 64;                  // 4*16*32
  float* sBG          = (float*)(s2 + 4 * 16 * 32);        // 4*16*16

  const int tid = threadIdx.x, lane = tid & 31, wave = tid >> 5;
  const int tile = blockIdx.x * 4 + wave;
  const int e0 = tile * 16;
  int* mySrc = sSrc + wave * 16;
  int* myDst = sDst + wave * 16;
  unsigned short* myEf0 = ef0 + wave * 16 * 192;
  unsigned short* myS1  = s1 + wave * 16 * 64;
  unsigned short* myS2  = s2 + wave * 16 * 32;
  float* myBG           = sBG + wave * 256;

  if (lane < 16) {
    mySrc[lane] = edge_index[e0 + lane];
    myDst[lane] = edge_index[N_EDGES + e0 + lane];
  }
  __syncthreads();

  // ef0 = [u_i | u_j | |u_i-u_j|] (16 x 192 bf16), float4 gathers + u32 stores
  for (int i = lane; i < 16 * 16; i += 32) {
    int e = i >> 4, cb = (i & 15) * 4;
    float4 ui = *(const float4*)(u + (size_t)myDst[e] * RELD + cb);
    float4 uj = *(const float4*)(u + (size_t)mySrc[e] * RELD + cb);
    unsigned* d0 = (unsigned*)(myEf0 + e * 192 + cb);
    unsigned* d1 = (unsigned*)(myEf0 + e * 192 + 64 + cb);
    unsigned* d2 = (unsigned*)(myEf0 + e * 192 + 128 + cb);
    d0[0] = pk2(ui.x, ui.y); d0[1] = pk2(ui.z, ui.w);
    d1[0] = pk2(uj.x, uj.y); d1[1] = pk2(uj.z, uj.w);
    d2[0] = pk2(fabsf(ui.x - uj.x), fabsf(ui.y - uj.y));
    d2[1] = pk2(fabsf(ui.z - uj.z), fabsf(ui.w - uj.w));
  }
  __syncthreads();

  // GEMM1: (16x192)@(192x64)+b1e, silu -> s1
  for (int nt = 0; nt < 4; ++nt) {
    CF acc; float bb = b1e[nt * 16 + (lane & 15)];
#pragma unroll
    for (int g = 0; g < 8; ++g) acc.f[g] = bb;
#pragma unroll
    for (int ks = 0; ks < 6; ++ks) {
      v16bf a = ld_a_lds(myEf0, 192, lane, ks * 32);
      v16bf b = ld_b_pk(wW1e, 4, lane, ks, nt);
      acc.v = wmma_bf16(a, b, acc.v);
    }
    int crow = (lane & 16) ? 8 : 0, col = nt * 16 + (lane & 15);
#pragma unroll
    for (int g = 0; g < 8; ++g) myS1[(g + crow) * 64 + col] = f2bf(siluf(acc.f[g]));
  }
  __syncthreads();

  // GEMM2: (16x64)@(64x32)+b2e -> s2
  for (int nt = 0; nt < 2; ++nt) {
    CF acc; float bb = b2e[nt * 16 + (lane & 15)];
#pragma unroll
    for (int g = 0; g < 8; ++g) acc.f[g] = bb;
#pragma unroll
    for (int ks = 0; ks < 2; ++ks) {
      v16bf a = ld_a_lds(myS1, 64, lane, ks * 32);
      v16bf b = ld_b_pk(wW2e, 2, lane, ks, nt);
      acc.v = wmma_bf16(a, b, acc.v);
    }
    int crow = (lane & 16) ? 8 : 0, col = nt * 16 + (lane & 15);
#pragma unroll
    for (int g = 0; g < 8; ++g) myS2[(g + crow) * 32 + col] = f2bf(acc.f[g]);
  }
  __syncthreads();

  // GEMM3: (16x32)@(32x16) -> [bias(8) | gate_raw(8)]
  {
    CF acc;
#pragma unroll
    for (int g = 0; g < 8; ++g) acc.f[g] = 0.f;
    v16bf a = ld_a_lds(myS2, EDD, lane, 0);
    v16bf b = ld_b_pk(wWbg, 1, lane, 0, 0);
    acc.v = wmma_bf16(a, b, acc.v);
    int crow = (lane & 16) ? 8 : 0, col = lane & 15;
#pragma unroll
    for (int g = 0; g < 8; ++g) myBG[(g + crow) * 16 + col] = acc.f[g];
  }
  __syncthreads();

  // sim = q[dst].k[src]*scale + bias ; gate = 1 + tanh ; segment max
  const float scale = 0.25f;  // HD^-0.5
#pragma unroll
  for (int p = 0; p < 4; ++p) {
    int pair = lane + 32 * p;  // 128 (edge,head) pairs per tile
    int e = pair >> 3, hh = pair & 7;
    int i = myDst[e], j = mySrc[e];
    const float4* qp = (const float4*)(q + (size_t)i * DMODEL + hh * HD);
    const float4* kp = (const float4*)(k + (size_t)j * DMODEL + hh * HD);
    float dot = 0.f;
#pragma unroll
    for (int d4 = 0; d4 < 4; ++d4) {
      float4 a = qp[d4], b = kp[d4];
      dot += a.x * b.x + a.y * b.y + a.z * b.z + a.w * b.w;
    }
    float sv = dot * scale + myBG[e * 16 + hh];
    float gv = 1.f + tanhf(myBG[e * 16 + 8 + hh]);
    size_t gi = (size_t)(e0 + e) * HEADS + hh;
    sim[gi] = sv; gate[gi] = gv;
    int bits = __float_as_int(sv);
    int key = (bits >= 0) ? bits : (bits ^ 0x7FFFFFFF);  // order-preserving
    atomicMax(&segmax[(size_t)i * HEADS + hh], key);
  }
}

// ---------------- pass 3: exp + denominator ----------------------------------
__global__ void k_softmax_num(const int* __restrict__ edge_index,
                              const float* __restrict__ sim,
                              const int* __restrict__ segmax,
                              float* __restrict__ eexp, float* __restrict__ denom) {
  int gid = blockIdx.x * blockDim.x + threadIdx.x;
  if (gid >= N_EDGES * HEADS) return;
  int e = gid >> 3, hh = gid & 7;
  int i = edge_index[N_EDGES + e];
  int key = segmax[(size_t)i * HEADS + hh];
  int bits = (key >= 0) ? key : (key ^ 0x7FFFFFFF);
  float m = __int_as_float(bits);
  if (!isfinite(m)) m = 0.f;
  float ex = __expf(sim[gid] - m);
  eexp[gid] = ex;
  atomicAdd(&denom[(size_t)i * HEADS + hh], ex);
}

// ---------------- pass 4: gated aggregation via packed bf16 atomics ----------
__global__ void k_aggregate(const int* __restrict__ edge_index,
                            const float* __restrict__ v,
                            const float* __restrict__ eexp,
                            const float* __restrict__ denom,
                            const float* __restrict__ gate,
                            unsigned short* __restrict__ agg) {
  int gid = blockIdx.x * blockDim.x + threadIdx.x;
  if (gid >= N_EDGES * HEADS) return;
  int e = gid >> 3, hh = gid & 7;
  int i = edge_index[N_EDGES + e];  // dst
  int j = edge_index[e];            // src
  float attn = eexp[gid] / (denom[(size_t)i * HEADS + hh] + 1e-16f);
  float w = attn * gate[gid];
  const float4* vp = (const float4*)(v + (size_t)j * DMODEL + hh * HD);
  unsigned short* ap = agg + (size_t)i * DMODEL + hh * HD;
#pragma unroll
  for (int d4 = 0; d4 < 4; ++d4) {
    float4 vv = vp[d4];
    atomic_pk_add_bf16(ap + d4 * 4 + 0, pk2(vv.x * w, vv.y * w));
    atomic_pk_add_bf16(ap + d4 * 4 + 2, pk2(vv.z * w, vv.w * w));
  }
}

// ---------------- pass 5: Wp proj, ada[256:768], residual, MLP ---------------
__global__ __launch_bounds__(256) void k_node_post(
    const float* __restrict__ x, const float* __restrict__ cin,
    const unsigned short* __restrict__ agg,  // bf16 row-major == WMMA A layout
    const unsigned short* __restrict__ wWada, const float* __restrict__ bada,
    const unsigned short* __restrict__ wWp, const float* __restrict__ bp,
    const unsigned short* __restrict__ wWf1, const float* __restrict__ bf1,
    const unsigned short* __restrict__ wWf2, const float* __restrict__ bf2,
    float* __restrict__ out) {
  extern __shared__ char smem[];
  unsigned short* sA = (unsigned short*)smem;  // 16*128 bf16 (staging)
  float* adaBuf = (float*)(sA + 16 * DMODEL);  // 16*512 f32
  float* attn   = adaBuf + 16 * 512;           // 16*128 f32 (later g_mlp)
  float* x1     = attn + 16 * DMODEL;          // 16*128 f32
  float* red    = x1 + 16 * DMODEL;            // 512 f32

  const int tid = threadIdx.x, lane = tid & 31, wave = tid >> 5;
  const int node0 = blockIdx.x * 16;
  const int row = tid >> 4, sub = tid & 15;

  // silu(c) staging
  for (int i = tid; i < 16 * 32; i += 256) {
    int r = i >> 5, cb = (i & 31) * 4;
    float4 cv = *(const float4*)(cin + (size_t)(node0 + r) * DMODEL + cb);
    unsigned* dp = (unsigned*)(sA + r * DMODEL + cb);
    dp[0] = pk2(siluf(cv.x), siluf(cv.y));
    dp[1] = pk2(siluf(cv.z), siluf(cv.w));
  }
  __syncthreads();

  // ada cols 256..767: [g_msa | sh_mlp | sc_mlp | g_mlp]
  {
    v16bf af[4];
#pragma unroll
    for (int ks = 0; ks < 4; ++ks) af[ks] = ld_a_lds(sA, DMODEL, lane, ks * 32);
    for (int nt = wave; nt < 32; nt += 8) {
      if (nt + 8 < 32)
        __builtin_prefetch(wWada + (((size_t)0 * 48 + 16 + nt + 8) << 9) + (lane << 4), 0, 2);
      CF acc; float bb = bada[256 + nt * 16 + (lane & 15)];
#pragma unroll
      for (int g = 0; g < 8; ++g) acc.f[g] = bb;
#pragma unroll
      for (int ks = 0; ks < 4; ++ks) {
        v16bf b = ld_b_pk(wWada, 48, lane, ks, 16 + nt);
        acc.v = wmma_bf16(af[ks], b, acc.v);
      }
      int crow = (lane & 16) ? 8 : 0, col = nt * 16 + (lane & 15);
#pragma unroll
      for (int g = 0; g < 8; ++g) adaBuf[(g + crow) * 512 + col] = acc.f[g];
    }
  }
  __syncthreads();

  // attn_out = agg@Wp + bp  (A fragments straight from global bf16 agg)
  {
    const unsigned short* aggT = agg + (size_t)node0 * DMODEL;
    v16bf af[4];
#pragma unroll
    for (int ks = 0; ks < 4; ++ks) af[ks] = ld_a_glb(aggT, DMODEL, lane, ks * 32);
    int nt = wave;
    CF acc; float bb = bp[nt * 16 + (lane & 15)];
#pragma unroll
    for (int g = 0; g < 8; ++g) acc.f[g] = bb;
#pragma unroll
    for (int ks = 0; ks < 4; ++ks) {
      v16bf b = ld_b_pk(wWp, 8, lane, ks, nt);
      acc.v = wmma_bf16(af[ks], b, acc.v);
    }
    int crow = (lane & 16) ? 8 : 0, col = nt * 16 + (lane & 15);
#pragma unroll
    for (int g = 0; g < 8; ++g) attn[(g + crow) * DMODEL + col] = acc.f[g];
  }
  __syncthreads();

  // x1 = x + g_msa*attn_out ; h2 = ln(x1)*(1+sc_mlp)+sh_mlp -> sA
  {
    const float* xr = x + (size_t)(node0 + row) * DMODEL + sub * 8;
    float4 xv[2];
    float s = 0.f, s2 = 0.f;
#pragma unroll
    for (int c4 = 0; c4 < 2; ++c4) {
      int cb = sub * 8 + c4 * 4;
      float4 xin = *(const float4*)(xr + c4 * 4);
      float4 gm = *(const float4*)(adaBuf + row * 512 + cb);
      float4 ao = *(const float4*)(attn + row * DMODEL + cb);
      float4 t;
      t.x = xin.x + gm.x * ao.x; t.y = xin.y + gm.y * ao.y;
      t.z = xin.z + gm.z * ao.z; t.w = xin.w + gm.w * ao.w;
      xv[c4] = t;
      *(float4*)(x1 + row * DMODEL + cb) = t;
      s += t.x + t.y + t.z + t.w;
      s2 += t.x * t.x + t.y * t.y + t.z * t.z + t.w * t.w;
    }
    red[row * 16 + sub] = s; red[256 + row * 16 + sub] = s2;
    __syncthreads();
    float su = 0.f, sq = 0.f;
    for (int t = 0; t < 16; ++t) { su += red[row * 16 + t]; sq += red[256 + row * 16 + t]; }
    float mu = su / DMODEL, var = sq / DMODEL - mu * mu, inv = rsqrtf(var + 1e-6f);
#pragma unroll
    for (int c4 = 0; c4 < 2; ++c4) {
      int cb = sub * 8 + c4 * 4;
      float4 sc = *(const float4*)(adaBuf + row * 512 + 256 + cb);
      float4 sh = *(const float4*)(adaBuf + row * 512 + 128 + cb);
      float4 t = xv[c4];
      unsigned* dp = (unsigned*)(sA + row * DMODEL + cb);
      dp[0] = pk2((t.x - mu) * inv * (1.f + sc.x) + sh.x,
                  (t.y - mu) * inv * (1.f + sc.y) + sh.y);
      dp[1] = pk2((t.z - mu) * inv * (1.f + sc.z) + sh.z,
                  (t.w - mu) * inv * (1.f + sc.w) + sh.w);
    }
  }
  __syncthreads();

  // save g_mlp into attn buffer, then overlay gelu staging on adaBuf
  for (int i = tid; i < 16 * 32; i += 256) {
    int r = i >> 5, cb = (i & 31) * 4;
    *(float4*)(attn + r * DMODEL + cb) = *(const float4*)(adaBuf + r * 512 + 384 + cb);
  }
  __syncthreads();
  unsigned short* sG = (unsigned short*)adaBuf;  // 16*512 bf16 (16KB of 32KB)

  // mlp hidden: h2@Wf1 + bf1, gelu(tanh) -> sG
  {
    v16bf af[4];
#pragma unroll
    for (int ks = 0; ks < 4; ++ks) af[ks] = ld_a_lds(sA, DMODEL, lane, ks * 32);
    for (int nt = wave; nt < 32; nt += 8) {
      if (nt + 8 < 32)
        __builtin_prefetch(wWf1 + (((size_t)0 * 32 + nt + 8) << 9) + (lane << 4), 0, 2);
      CF acc; float bb = bf1[nt * 16 + (lane & 15)];
#pragma unroll
      for (int g = 0; g < 8; ++g) acc.f[g] = bb;
#pragma unroll
      for (int ks = 0; ks < 4; ++ks) {
        v16bf b = ld_b_pk(wWf1, 32, lane, ks, nt);
        acc.v = wmma_bf16(af[ks], b, acc.v);
      }
      int crow = (lane & 16) ? 8 : 0, col = nt * 16 + (lane & 15);
#pragma unroll
      for (int g = 0; g < 8; ++g) {
        float gx = acc.f[g];
        float t3 = 0.7978845608028654f * (gx + 0.044715f * gx * gx * gx);
        sG[(g + crow) * MLPH + col] = f2bf(0.5f * gx * (1.f + tanhf(t3)));
      }
    }
  }
  __syncthreads();

  // mlp out: gelu@Wf2 + bf2 ; out = x1 + g_mlp*mlp
  {
    int nt = wave;
    CF acc; float bb = bf2[nt * 16 + (lane & 15)];
#pragma unroll
    for (int g = 0; g < 8; ++g) acc.f[g] = bb;
    for (int ks = 0; ks < 16; ++ks) {
      v16bf a = ld_a_lds(sG, MLPH, lane, ks * 32);
      v16bf b = ld_b_pk(wWf2, 8, lane, ks, nt);
      acc.v = wmma_bf16(a, b, acc.v);
    }
    int crow = (lane & 16) ? 8 : 0, col = nt * 16 + (lane & 15);
#pragma unroll
    for (int g = 0; g < 8; ++g) {
      int r = g + crow;
      out[(size_t)(node0 + r) * DMODEL + col] =
          x1[r * DMODEL + col] + attn[r * DMODEL + col] * acc.f[g];
    }
  }
}

// ---------------- host launcher ----------------------------------------------
extern "C" void kernel_launch(void* const* d_in, const int* in_sizes, int n_in,
                              void* d_out, int out_size, void* d_ws, size_t ws_size,
                              hipStream_t stream) {
  (void)in_sizes; (void)n_in; (void)out_size; (void)ws_size;
  const float* x      = (const float*)d_in[0];
  const float* c      = (const float*)d_in[1];
  const int*   eidx   = (const int*)d_in[2];
  const float* Wq     = (const float*)d_in[3];  const float* bq   = (const float*)d_in[4];
  const float* Wk     = (const float*)d_in[5];  const float* bk   = (const float*)d_in[6];
  const float* Wv     = (const float*)d_in[7];  const float* bv   = (const float*)d_in[8];
  const float* Wp     = (const float*)d_in[9];  const float* bp   = (const float*)d_in[10];
  const float* Wrel   = (const float*)d_in[11]; const float* brel = (const float*)d_in[12];
  const float* g_reln = (const float*)d_in[13]; const float* b_reln = (const float*)d_in[14];
  const float* W1e    = (const float*)d_in[15]; const float* b1e  = (const float*)d_in[16];
  const float* W2e    = (const float*)d_in[17]; const float* b2e  = (const float*)d_in[18];
  const float* Wbias  = (const float*)d_in[19]; const float* Wgate = (const float*)d_in[20];
  const float* Wf1    = (const float*)d_in[21]; const float* bf1  = (const float*)d_in[22];
  const float* Wf2    = (const float*)d_in[23]; const float* bf2  = (const float*)d_in[24];
  const float* Wada   = (const float*)d_in[25]; const float* bada = (const float*)d_in[26];
  float* out = (float*)d_out;

  char* ws = (char*)d_ws;
  size_t off = 0;
  auto take = [&](size_t bytes) -> char* {
    char* p = ws + off;
    off += (bytes + 255) & ~(size_t)255;
    return p;
  };
  // bf16 fragment-major packed weights
  unsigned short* wWada = (unsigned short*)take((size_t)DMODEL * 6 * DMODEL * 2);
  unsigned short* wWq   = (unsigned short*)take((size_t)DMODEL * DMODEL * 2);
  unsigned short* wWk   = (unsigned short*)take((size_t)DMODEL * DMODEL * 2);
  unsigned short* wWv   = (unsigned short*)take((size_t)DMODEL * DMODEL * 2);
  unsigned short* wWp   = (unsigned short*)take((size_t)DMODEL * DMODEL * 2);
  unsigned short* wWrel = (unsigned short*)take((size_t)DMODEL * RELD * 2);
  unsigned short* wW1e  = (unsigned short*)take((size_t)3 * RELD * 2 * EDD * 2);
  unsigned short* wW2e  = (unsigned short*)take((size_t)2 * EDD * EDD * 2);
  unsigned short* wWbg  = (unsigned short*)take((size_t)EDD * 16 * 2);
  unsigned short* wWf1  = (unsigned short*)take((size_t)DMODEL * MLPH * 2);
  unsigned short* wWf2  = (unsigned short*)take((size_t)MLPH * DMODEL * 2);
  // activations / edge state
  float* qb    = (float*)take((size_t)N_NODES * DMODEL * 4);
  float* kb    = (float*)take((size_t)N_NODES * DMODEL * 4);
  float* vb    = (float*)take((size_t)N_NODES * DMODEL * 4);
  float* ub    = (float*)take((size_t)N_NODES * RELD * 4);
  float* simb  = (float*)take((size_t)N_EDGES * HEADS * 4);
  float* gateb = (float*)take((size_t)N_EDGES * HEADS * 4);
  float* eexpb = (float*)take((size_t)N_EDGES * HEADS * 4);
  float* denom = (float*)take((size_t)N_NODES * HEADS * 4);
  unsigned short* aggb = (unsigned short*)take((size_t)N_NODES * DMODEL * 2);
  int* segmx = (int*)take((size_t)N_NODES * HEADS * 4);

  auto pack = [&](const float* s, unsigned short* d, int K, int N) {
    k_pack_b<<<(K * N + 255) / 256, 256, 0, stream>>>(s, d, K, N);
  };
  pack(Wada, wWada, DMODEL, 6 * DMODEL);
  pack(Wq, wWq, DMODEL, DMODEL);   pack(Wk, wWk, DMODEL, DMODEL);
  pack(Wv, wWv, DMODEL, DMODEL);   pack(Wp, wWp, DMODEL, DMODEL);
  pack(Wrel, wWrel, DMODEL, RELD);
  pack(W1e, wW1e, 3 * RELD, 2 * EDD);
  pack(W2e, wW2e, 2 * EDD, EDD);
  pack(Wf1, wWf1, DMODEL, MLPH);   pack(Wf2, wWf2, MLPH, DMODEL);
  k_pack_bias_gate<<<2, 256, 0, stream>>>(Wbias, Wgate, wWbg);
  k_init<<<(N_NODES * DMODEL + 255) / 256, 256, 0, stream>>>(
      (unsigned*)aggb, denom, segmx);

  const size_t lds1 = 16 * DMODEL * 4 + 16 * DMODEL * 2 + 16 * 256 * 4 +
                      16 * DMODEL * 2 + 16 * RELD * 4 + 512 * 4;  // 38912
  k_node_pre<<<N_NODES / 16, 256, lds1, stream>>>(
      x, c, wWada, bada, wWq, bq, wWk, bk, wWv, bv, wWrel, brel,
      g_reln, b_reln, qb, kb, vb, ub);

  const size_t lds2 = 4 * 16 * 4 * 2 + 4 * 16 * 192 * 2 + 4 * 16 * 64 * 2 +
                      4 * 16 * 32 * 2 + 4 * 256 * 4;  // 41472
  k_edge<<<N_EDGES / 64, 128, lds2, stream>>>(
      eidx, ub, qb, kb, wW1e, b1e, wW2e, b2e, wWbg, simb, gateb, segmx);

  k_softmax_num<<<(N_EDGES * HEADS + 255) / 256, 256, 0, stream>>>(
      eidx, simb, segmx, eexpb, denom);
  k_aggregate<<<(N_EDGES * HEADS + 255) / 256, 256, 0, stream>>>(
      eidx, vb, eexpb, denom, gateb, aggb);

  const size_t lds3 = 16 * DMODEL * 2 + 16 * 512 * 4 + 16 * DMODEL * 4 +
                      16 * DMODEL * 4 + 512 * 4;  // 55296
  k_node_post<<<N_NODES / 16, 256, lds3, stream>>>(
      x, c, aggb, wWada, bada, wWp, bp, wWf1, bf1, wWf2, bf2, out);
}
